// MultiRelationGNN_75746043232940
// MI455X (gfx1250) — matrix operand order repaired and checked by reading, not compiled, not run on hardware
//
#include <hip/hip_runtime.h>

#define IN_DIM 128
#define H_DIM  64
#define RR     5
#define NEG_SLOPE 0.01f
#define PCOLS  (RR * 2 * H_DIM)   // 640

typedef __attribute__((ext_vector_type(16))) __bf16 v16bf;
typedef __attribute__((ext_vector_type(8)))  float  v8f;

// ---------- WMMA fragment helpers (wave32, 16x16x32 bf16, per ISA 7.12.2) ----

// A fragment from row-major f32 (lda = row stride), vectorized float4 loads.
// lane l<16 : row m0+l,   K = k0 + {0..15}
// lane l>=16: row m0+l-16, K = k0 + {16..31}
static __device__ __forceinline__ v16bf load_a_bf16(const float* A, int lda, int m0, int k0) {
  const int lane = threadIdx.x & 31;
  const int hf   = lane >> 4;
  const float4* base =
      (const float4*)(A + (size_t)(m0 + (lane & 15)) * lda + k0 + hf * 16);
  const float4 q0 = base[0], q1 = base[1];   // K = kb+0..7
  const float4 q2 = base[2], q3 = base[3];   // K = kb+8..15
  v16bf a;
  a[0]  = (__bf16)q0.x; a[1]  = (__bf16)q0.y; a[2]  = (__bf16)q0.z; a[3]  = (__bf16)q0.w;
  a[4]  = (__bf16)q1.x; a[5]  = (__bf16)q1.y; a[6]  = (__bf16)q1.z; a[7]  = (__bf16)q1.w;
  a[8]  = (__bf16)q2.x; a[9]  = (__bf16)q2.y; a[10] = (__bf16)q2.z; a[11] = (__bf16)q2.w;
  a[12] = (__bf16)q3.x; a[13] = (__bf16)q3.y; a[14] = (__bf16)q3.z; a[15] = (__bf16)q3.w;
  return a;
}

// B fragment from pre-packed transposed bf16 weights BT[n][k] (ldk = K stride).
// lane l: col n0+(l&15); K = k0 + (l>=16 ? 16 : 0) + j, j=0..15 -> 32 contiguous bytes
static __device__ __forceinline__ v16bf load_bT(const __bf16* BT, int ldk, int k0, int n0) {
  const int lane = threadIdx.x & 31;
  const int hf   = lane >> 4;
  return *(const v16bf*)(BT + (size_t)(n0 + (lane & 15)) * ldk + k0 + hf * 16);
}

// C/D store: v8f -> row-major f32, rows m0 + hf*8 + i, col n0 + (lane&15)
static __device__ __forceinline__ void store_c(float* C, int ldc, int m0, int n0, v8f acc) {
  const int lane = threadIdx.x & 31;
  const int hf   = lane >> 4;
  float* base = C + (size_t)(m0 + hf * 8) * ldc + n0 + (lane & 15);
#pragma unroll
  for (int i = 0; i < 8; ++i) base[(size_t)i * ldc] = acc[i];
}

#define WMMA_BF16(a, b, c) \
  __builtin_amdgcn_wmma_f32_16x16x32_bf16(false, (a), false, (b), (short)0, (c), false, false)

// ---------------------------- packing kernels --------------------------------

__global__ void zero_f32(float* __restrict__ p, long n) {
  long i = (long)blockIdx.x * blockDim.x + threadIdx.x;
  if (i < n) p[i] = 0.0f;
}

// dst[n*K + k] = (bf16) src[k*N + n]    (src is K x N row-major)
__global__ void transpose_to_bf16(const float* __restrict__ src, __bf16* __restrict__ dst,
                                  int K, int N) {
  int i = blockIdx.x * blockDim.x + threadIdx.x;
  if (i >= K * N) return;
  int n = i / K, k = i - n * K;
  dst[i] = (__bf16)src[(size_t)k * N + n];
}

// WcatT[c*64 + k] = relW[r][bot*64 + k][o]; c -> r=c>>7, bot=(c>>6)&1, o=c&63
__global__ void pack_relW(const float* __restrict__ relW, __bf16* __restrict__ dst) {
  int i = blockIdx.x * blockDim.x + threadIdx.x;   // PCOLS*64 elements
  if (i >= PCOLS * H_DIM) return;
  int c = i >> 6, k = i & 63;
  int r = c >> 7, bot = (c >> 6) & 1, o = c & 63;
  dst[i] = (__bf16)relW[((size_t)r * 128 + bot * 64 + k) * 64 + o];
}

// ---------------------------- compute kernels --------------------------------

// emb0[N,64] = X[N,128] @ Wf[128,64] + bf     (one wave per 16-row strip)
__global__ void __launch_bounds__(128)
field_gemm(const float* __restrict__ X, const __bf16* __restrict__ WfT,
           const float* __restrict__ bf, float* __restrict__ emb0, int Mt) {
  const int mt = blockIdx.x * 4 + (threadIdx.x >> 5);
  if (mt >= Mt) return;
  const int m0 = mt * 16;
  v8f acc[4] = {};
#pragma unroll
  for (int ks = 0; ks < 4; ++ks) {
    v16bf a = load_a_bf16(X, IN_DIM, m0, ks * 32);
#pragma unroll
    for (int nt = 0; nt < 4; ++nt) {
      v16bf b = load_bT(WfT, IN_DIM, ks * 32, nt * 16);
      acc[nt] = WMMA_BF16(a, b, acc[nt]);
    }
  }
  const int lane = threadIdx.x & 31;
#pragma unroll
  for (int nt = 0; nt < 4; ++nt) {
    float bias = bf[nt * 16 + (lane & 15)];
#pragma unroll
    for (int i = 0; i < 8; ++i) acc[nt][i] += bias;
    store_c(emb0, H_DIM, m0, nt * 16, acc[nt]);
  }
}

// P[N,640] = H[N,64] @ Wcat[64,640]; grid (ceil(Mt/4), 10), 16x64 tile per wave
__global__ void __launch_bounds__(128)
noderel_gemm(const float* __restrict__ H, const __bf16* __restrict__ WcatT,
             float* __restrict__ P, int Mt) {
  const int mt = blockIdx.x * 4 + (threadIdx.x >> 5);
  if (mt >= Mt) return;
  const int m0 = mt * 16;
  const int c0 = blockIdx.y * 64;
  v8f acc[4] = {};
#pragma unroll
  for (int ks = 0; ks < 2; ++ks) {
    v16bf a = load_a_bf16(H, H_DIM, m0, ks * 32);
#pragma unroll
    for (int nt = 0; nt < 4; ++nt) {
      v16bf b = load_bT(WcatT, H_DIM, ks * 32, c0 + nt * 16);
      acc[nt] = WMMA_BF16(a, b, acc[nt]);
    }
  }
#pragma unroll
  for (int nt = 0; nt < 4; ++nt) store_c(P, PCOLS, m0, c0 + nt * 16, acc[nt]);
}

// 2 edges per wave, 16 lanes/edge, float4/lane:
// out[dst] += w * (P[src,r,0:64] + P[dst,r,64:128] + b[r])
__global__ void __launch_bounds__(256)
edge_pass(const int* __restrict__ ei, const int* __restrict__ et,
          const float* __restrict__ etime, const float* __restrict__ P,
          const float* __restrict__ relb, const float* __restrict__ plam,
          const float* __restrict__ pbeta, float* __restrict__ outacc, int E) {
  const long tid = blockIdx.x * (long)blockDim.x + threadIdx.x;
  const int e = (int)(tid >> 4);
  if (e >= E) return;
  const int q = threadIdx.x & 15;          // float4 slot within the 64-float row
  const int src = ei[e];
  const int dst = ei[E + e];
  const int r   = et[e];
  const float w = plam[0] * __expf(-pbeta[0] * fabsf(etime[e]));
  const float4* pt = (const float4*)(P + (size_t)src * PCOLS + r * 128);
  const float4* pb = (const float4*)(P + (size_t)dst * PCOLS + r * 128 + 64);
  const float4* bb = (const float4*)(relb + r * H_DIM);
  const float4 a = pt[q], b = pb[q], c = bb[q];
  float* o = outacc + (size_t)dst * H_DIM + q * 4;
  atomicAdd(o + 0, w * (a.x + b.x + c.x));
  atomicAdd(o + 1, w * (a.y + b.y + c.y));
  atomicAdd(o + 2, w * (a.z + b.z + c.z));
  atomicAdd(o + 3, w * (a.w + b.w + c.w));
}

// out = lrelu(e2@W2+b2) + lrelu(e1@W1+b1) + lrelu(e0@W0+b0)
__global__ void __launch_bounds__(128)
out_mlp(const float* __restrict__ e0, const float* __restrict__ e1,
        const float* __restrict__ e2,
        const __bf16* __restrict__ W0T, const float* __restrict__ b0,
        const __bf16* __restrict__ W1T, const float* __restrict__ b1,
        const __bf16* __restrict__ W2T, const float* __restrict__ b2,
        float* __restrict__ out, int Mt) {
  const int mt = blockIdx.x * 4 + (threadIdx.x >> 5);
  if (mt >= Mt) return;
  const int m0 = mt * 16;
  const int lane = threadIdx.x & 31;
  v16bf a0[2], a1[2], a2[2];
#pragma unroll
  for (int ks = 0; ks < 2; ++ks) {
    a0[ks] = load_a_bf16(e0, H_DIM, m0, ks * 32);
    a1[ks] = load_a_bf16(e1, H_DIM, m0, ks * 32);
    a2[ks] = load_a_bf16(e2, H_DIM, m0, ks * 32);
  }
#pragma unroll
  for (int nt = 0; nt < 4; ++nt) {
    v8f z0 = {}, z1 = {}, z2 = {};
#pragma unroll
    for (int ks = 0; ks < 2; ++ks) {
      z0 = WMMA_BF16(a0[ks], load_bT(W0T, H_DIM, ks * 32, nt * 16), z0);
      z1 = WMMA_BF16(a1[ks], load_bT(W1T, H_DIM, ks * 32, nt * 16), z1);
      z2 = WMMA_BF16(a2[ks], load_bT(W2T, H_DIM, ks * 32, nt * 16), z2);
    }
    const int col = nt * 16 + (lane & 15);
    const float bi0 = b0[col], bi1 = b1[col], bi2 = b2[col];
    v8f res;
#pragma unroll
    for (int i = 0; i < 8; ++i) {
      float v0 = z0[i] + bi0; v0 = v0 >= 0.0f ? v0 : NEG_SLOPE * v0;
      float v1 = z1[i] + bi1; v1 = v1 >= 0.0f ? v1 : NEG_SLOPE * v1;
      float v2 = z2[i] + bi2; v2 = v2 >= 0.0f ? v2 : NEG_SLOPE * v2;
      res[i] = v0 + v1 + v2;
    }
    store_c(out, H_DIM, m0, nt * 16, res);
  }
}

// ---------------------------- launcher --------------------------------------

extern "C" void kernel_launch(void* const* d_in, const int* in_sizes, int n_in,
                              void* d_out, int out_size, void* d_ws, size_t ws_size,
                              hipStream_t stream) {
  (void)n_in; (void)out_size; (void)ws_size;
  const float* x      = (const float*)d_in[0];
  const int*   ei     = (const int*)  d_in[1];
  const int*   et     = (const int*)  d_in[2];
  const float* etime  = (const float*)d_in[3];
  const float* fieldW = (const float*)d_in[4];
  const float* fieldb = (const float*)d_in[5];
  const float* relW1  = (const float*)d_in[6];
  const float* relb1  = (const float*)d_in[7];
  const float* relW2  = (const float*)d_in[8];
  const float* relb2  = (const float*)d_in[9];
  const float* outW0  = (const float*)d_in[10];
  const float* outb0  = (const float*)d_in[11];
  const float* outW1  = (const float*)d_in[12];
  const float* outb1  = (const float*)d_in[13];
  const float* outW2  = (const float*)d_in[14];
  const float* outb2  = (const float*)d_in[15];
  const float* lam    = (const float*)d_in[16];
  const float* beta   = (const float*)d_in[17];

  const int N  = in_sizes[0] / IN_DIM;   // 50000
  const int E  = in_sizes[2];            // 1600000
  const int Mt = (N + 15) / 16;          // 3125

  // workspace layout: emb0 | emb1 | emb2 | P (f32), then bf16 packed weights
  float* emb0 = (float*)d_ws;
  float* emb1 = emb0 + (size_t)N * H_DIM;
  float* emb2 = emb1 + (size_t)N * H_DIM;
  float* P    = emb2 + (size_t)N * H_DIM;          // N x 640
  __bf16* WfT    = (__bf16*)(P + (size_t)N * PCOLS);
  __bf16* Wcat1T = WfT    + (size_t)H_DIM * IN_DIM;    //  64x128
  __bf16* Wcat2T = Wcat1T + (size_t)PCOLS * H_DIM;     // 640x64
  __bf16* W0T    = Wcat2T + (size_t)PCOLS * H_DIM;
  __bf16* W1T    = W0T    + (size_t)H_DIM * H_DIM;
  __bf16* W2T    = W1T    + (size_t)H_DIM * H_DIM;

  // pack weights (bf16, transposed) — pure function of inputs, runs every call
  transpose_to_bf16<<<(IN_DIM * H_DIM + 255) / 256, 256, 0, stream>>>(fieldW, WfT, IN_DIM, H_DIM);
  pack_relW<<<(PCOLS * H_DIM + 255) / 256, 256, 0, stream>>>(relW1, Wcat1T);
  pack_relW<<<(PCOLS * H_DIM + 255) / 256, 256, 0, stream>>>(relW2, Wcat2T);
  transpose_to_bf16<<<(H_DIM * H_DIM + 255) / 256, 256, 0, stream>>>(outW0, W0T, H_DIM, H_DIM);
  transpose_to_bf16<<<(H_DIM * H_DIM + 255) / 256, 256, 0, stream>>>(outW1, W1T, H_DIM, H_DIM);
  transpose_to_bf16<<<(H_DIM * H_DIM + 255) / 256, 256, 0, stream>>>(outW2, W2T, H_DIM, H_DIM);

  // zero the two atomic accumulators (emb1, emb2 contiguous)
  const long nz = 2L * N * H_DIM;
  zero_f32<<<(int)((nz + 255) / 256), 256, 0, stream>>>(emb1, nz);

  const int gb = (Mt + 3) / 4;        // 4 waves (16-row tiles) per 128-thread block
  const int ge = (E * 16 + 255) / 256;  // 16 threads per edge

  field_gemm<<<gb, 128, 0, stream>>>(x, WfT, fieldb, emb0, Mt);

  noderel_gemm<<<dim3(gb, 10), 128, 0, stream>>>(emb0, Wcat1T, P, Mt);
  edge_pass<<<ge, 256, 0, stream>>>(ei, et, etime, P, relb1, lam, beta, emb1, E);

  noderel_gemm<<<dim3(gb, 10), 128, 0, stream>>>(emb1, Wcat2T, P, Mt);
  edge_pass<<<ge, 256, 0, stream>>>(ei, et, etime, P, relb2, lam, beta, emb2, E);

  out_mlp<<<gb, 128, 0, stream>>>(emb0, emb1, emb2, W0T, outb0, W1T, outb1,
                                  W2T, outb2, (float*)d_out, Mt);
}
